// DLRMv2_59167469469771
// MI455X (gfx1250) — matrix operand order repaired and checked by reading, not compiled, not run on hardware
//
#include <hip/hip_runtime.h>
#include <hip/hip_bf16.h>
#include <math.h>

// ---------------------------------------------------------------------------
// DLRM-v2 style pipeline for gfx1250 (MI455X), wave32 + WMMA bf16.
//  B=8192, P=4, V=100000, D=128, NS=26, ND=13
//  HBM floor: 436MB embedding gather (~19us @23.3TB/s). GEMMs (~10.5 GFLOP)
//  run on v_wmma_f32_16x16x32_bf16 with BOTH operands pre-swizzled into the
//  ISA register layouts, so the WMMA loop is pure loads + wmma.
// ---------------------------------------------------------------------------

typedef __bf16        v16bf __attribute__((ext_vector_type(16)));
typedef float         v8f   __attribute__((ext_vector_type(8)));
typedef unsigned int  v8u   __attribute__((ext_vector_type(8)));
typedef unsigned int  u32;

static constexpr int kB  = 8192;
static constexpr int kP  = 4;
static constexpr int kV  = 100000;
static constexpr int kD  = 128;
static constexpr int kNS = 26;
static constexpr int kND = 13;
static constexpr int kXdim = 480;   // 128 + 351 padded to 480 (15 K-chunks)
static constexpr int kMt  = kB / 16;  // 512 M-tiles

// round-to-nearest-even f32 -> bf16, packed pair into one dword
__device__ __forceinline__ u32 bf16pk(float a, float b) {
  u32 ua = __float_as_uint(a); ua = (ua + 0x7FFFu + ((ua >> 16) & 1u)) >> 16;
  u32 ub = __float_as_uint(b); ub = (ub + 0x7FFFu + ((ub >> 16) & 1u)) >> 16;
  return (ua & 0xFFFFu) | (ub << 16);
}

// ---------------------------------------------------------------------------
// Weight pre-swizzle: W[K x N] fp32 -> bf16 in WMMA B-matrix register layout.
//   dword i = ((ntile*nKc + kc)*32 + lane)*8 + vgpr
//   lane L: n = ntile*16 + L%16 ; K = kc*32 + (vgpr/4)*16 + (L/16)*8 + (vgpr%4)*2 (+1 hi16)
// ---------------------------------------------------------------------------
__global__ void pack_w(const float* __restrict__ W, u32* __restrict__ out,
                       int K, int N, int nKc) {
  int i = blockIdx.x * blockDim.x + threadIdx.x;
  if (i >= nKc * (N >> 4) * 256) return;
  int vg   = i & 7;
  int lane = (i >> 3) & 31;
  int kc   = (i >> 8) % nKc;
  int nt   = (i >> 8) / nKc;
  int n  = nt * 16 + (lane & 15);
  int k0 = kc * 32 + ((vg >> 2) << 4) + ((lane >> 4) << 3) + (vg & 3) * 2;
  float f0 = (k0     < K) ? W[(size_t)k0       * N + n] : 0.f;
  float f1 = (k0 + 1 < K) ? W[(size_t)(k0 + 1) * N + n] : 0.f;
  out[i] = bf16pk(f0, f1);
}

// ---------------------------------------------------------------------------
// Activation pre-swizzle: A[M x lda] fp32 -> bf16 in WMMA A-matrix layout.
//   dword i = ((mtile*nKc + kc)*32 + lane)*8 + vgpr
//   lane L: m = mtile*16 + L%16
//   K = kc*32 + (vgpr/4)*16 + (L/16)*8 + (vgpr%4)*2  (+1 in hi16)
// ---------------------------------------------------------------------------
__global__ void pack_a(const float* __restrict__ A, u32* __restrict__ out,
                       int lda, int K, int nKc) {
  int i = blockIdx.x * blockDim.x + threadIdx.x;
  if (i >= kMt * nKc * 256) return;
  int vg   = i & 7;
  int lane = (i >> 3) & 31;
  int kc   = (i >> 8) % nKc;
  int mt   = (i >> 8) / nKc;
  int row = mt * 16 + (lane & 15);
  int k0  = kc * 32 + ((vg >> 2) << 4) + ((lane >> 4) << 3) + (vg & 3) * 2;
  const float* ar = A + (size_t)row * lda;
  float f0 = (k0     < K) ? ar[k0]     : 0.f;
  float f1 = (k0 + 1 < K) ? ar[k0 + 1] : 0.f;
  out[i] = bf16pk(f0, f1);
}

// ---------------------------------------------------------------------------
// GEMM on pre-swizzled operands: C = relu(A*W + bias).
// Block = 256 (8 waves); wave owns a 16(M) x 64(N) strip; grid (N/64, M/128).
// Inner loop per K-chunk: 1 v8u A load + 4 v8u B loads + 4 WMMAs. Epilogue
// optionally writes fp32 row-major (Cf32) and/or the NEXT layer's packed-A
// (ApOut): strip -> per-wave LDS bf16 -> A-layout readback (no __syncthreads
// needed, regions are wave-private).
// ---------------------------------------------------------------------------
__global__ __launch_bounds__(256)
void gemm_ws(const u32* __restrict__ Ap, int nKc,
             const u32* __restrict__ Wp, const float* __restrict__ bias,
             float* __restrict__ Cf32, int N,
             u32* __restrict__ ApOut, int nKcOut) {
  __shared__ __bf16 S[8][16][64];           // 16 KB strip-repack staging
  const int lane = threadIdx.x & 31;
  const int w    = threadIdx.x >> 5;
  const int half = lane >> 4;
  const int lm   = lane & 15;
  const int m0   = blockIdx.y * 128 + w * 16;
  const int mt   = m0 >> 4;
  const int n0   = blockIdx.x * 64;
  const int nt0  = n0 >> 4;

  v8f acc[4] = {{}, {}, {}, {}};
  const v8u* aP = reinterpret_cast<const v8u*>(Ap) + (size_t)mt * nKc * 32 + lane;
  const v8u* bP[4];
#pragma unroll
  for (int t = 0; t < 4; ++t)
    bP[t] = reinterpret_cast<const v8u*>(Wp) + (size_t)(nt0 + t) * nKc * 32 + lane;

  for (int kc = 0; kc < nKc; ++kc) {
    v16bf aT = __builtin_bit_cast(v16bf, aP[kc * 32]);
#pragma unroll
    for (int t = 0; t < 4; ++t) {
      v16bf bT = __builtin_bit_cast(v16bf, bP[t][kc * 32]);
      acc[t] = __builtin_amdgcn_wmma_f32_16x16x32_bf16(
          false, aT, false, bT, (short)0, acc[t], false, false);
    }
  }

  // Epilogue: bias + relu; C layout VGPR r -> row m0 + r + 8*half
#pragma unroll
  for (int t = 0; t < 4; ++t) {
    int n = n0 + t * 16 + lm;
    float bv = bias[n];
#pragma unroll
    for (int r = 0; r < 8; ++r) {
      float v = acc[t][r] + bv;
      v = v > 0.f ? v : 0.f;
      if (Cf32) Cf32[(size_t)(m0 + r + 8 * half) * N + n] = v;
      if (ApOut) S[w][r + 8 * half][t * 16 + lm] = (__bf16)v;
    }
  }
  if (ApOut) {
    // Wave strip (16 rows x 64 cols) == next layer A-tiles (mt, n0/32 + {0,1})
    const u32* S32 = reinterpret_cast<const u32*>(&S[w][0][0]);  // [16][32]
#pragma unroll
    for (int ch = 0; ch < 2; ++ch) {
      int d0 = lm * 32 + ch * 16 + half * 4;
      const uint4* q = reinterpret_cast<const uint4*>(&S32[d0]);
      uint4 lo = q[0];
      uint4 hi = *reinterpret_cast<const uint4*>(&S32[d0 + 8]);
      v8u av = {lo.x, lo.y, lo.z, lo.w, hi.x, hi.y, hi.z, hi.w};
      size_t o = ((size_t)mt * nKcOut + ((n0 >> 5) + ch)) * 32 + lane;
      reinterpret_cast<v8u*>(ApOut)[o] = av;
    }
  }
}

// ---------------------------------------------------------------------------
// Embedding bag pooling: one wave per (batch, table); sum P=4 rows of 128
// fp32; store packed bf16 pairs. grid = B*NS/8.
// ---------------------------------------------------------------------------
__global__ __launch_bounds__(256)
void emb_pool(const int* __restrict__ sx, const float* __restrict__ tables,
              u32* __restrict__ pooled) {
  const int lane = threadIdx.x & 31;
  const int w    = threadIdx.x >> 5;
  const long g = (long)blockIdx.x * 8 + w;
  const int b = (int)(g / kNS);
  const int s = (int)(g % kNS);
  const float* tb = tables + (size_t)s * kV * kD;
  float4 sum = {0.f, 0.f, 0.f, 0.f};
#pragma unroll
  for (int p = 0; p < kP; ++p) {
    int idx = sx[(size_t)(b * kP + p) * kNS + s];
    float4 v = reinterpret_cast<const float4*>(tb + (size_t)idx * kD)[lane];
    sum.x += v.x; sum.y += v.y; sum.z += v.z; sum.w += v.w;
  }
  size_t o = ((size_t)b * kNS + s) * 64 + lane * 2;
  pooled[o]     = bf16pk(sum.x, sum.y);
  pooled[o + 1] = bf16pk(sum.z, sum.w);
}

// ---------------------------------------------------------------------------
// Feature interaction: per batch row, G = E * E^T, E = [dense_out; pooled]
// (27 x 128 zero-padded to 32 x 128 bf16 in LDS). 8 rows/block, 1 wave/row,
// 16 WMMAs each. Emits x[b] = [dense fp32 | upper-tri(G) | 0-pad] (480).
// ---------------------------------------------------------------------------
__global__ __launch_bounds__(256)
void interact(const float* __restrict__ dense_out, const u32* __restrict__ pooled,
              float* __restrict__ x) {
  __shared__ u32 E[8 * 32 * 64];   // 64 KB
  const int tid = threadIdx.x;
  const int bbase = blockIdx.x * 8;

  for (int i = tid; i < 8 * 32 * 64; i += 256) {
    int bl = i >> 11, er = (i >> 6) & 31, dc = i & 63;
    int b = bbase + bl;
    u32 v = 0;
    if (er == 0)
      v = bf16pk(dense_out[(size_t)b * kD + 2 * dc],
                 dense_out[(size_t)b * kD + 2 * dc + 1]);
    else if (er < 27)
      v = pooled[((size_t)b * kNS + (er - 1)) * 64 + dc];
    E[i] = v;
  }
  __syncthreads();

  const int lane = tid & 31;
  const int w    = tid >> 5;
  const int half = lane >> 4;
  const int lm   = lane & 15;
  const int b    = bbase + w;
  u32* Ew = &E[w * 2048];

  v8f acc[2][2] = {{{}, {}}, {{}, {}}};
#pragma unroll
  for (int kc = 0; kc < 4; ++kc) {
    v16bf At[2], Bt[2];
#pragma unroll
    for (int mt = 0; mt < 2; ++mt) {
      int r = mt * 16 + lm;
      uint4 lo = *reinterpret_cast<const uint4*>(&Ew[r * 64 + kc * 16 + half * 4]);
      uint4 hi = *reinterpret_cast<const uint4*>(&Ew[r * 64 + kc * 16 + 8 + half * 4]);
      v8u av = {lo.x, lo.y, lo.z, lo.w, hi.x, hi.y, hi.z, hi.w};
      At[mt] = __builtin_bit_cast(v16bf, av);
    }
#pragma unroll
    for (int nt = 0; nt < 2; ++nt) {
      int r = nt * 16 + lm;
      const uint4* p0 = reinterpret_cast<const uint4*>(&Ew[r * 64 + kc * 16 + half * 4]);
      uint4 lo = p0[0], hi = p0[2];
      v8u bv = {lo.x, lo.y, lo.z, lo.w, hi.x, hi.y, hi.z, hi.w};
      Bt[nt] = __builtin_bit_cast(v16bf, bv);
    }
#pragma unroll
    for (int mt = 0; mt < 2; ++mt)
#pragma unroll
      for (int nt = 0; nt < 2; ++nt)
        acc[mt][nt] = __builtin_amdgcn_wmma_f32_16x16x32_bf16(
            false, At[mt], false, Bt[nt], (short)0, acc[mt][nt], false, false);
  }

  float* G = reinterpret_cast<float*>(Ew);       // reuse wave-private region
#pragma unroll
  for (int mt = 0; mt < 2; ++mt)
#pragma unroll
    for (int nt = 0; nt < 2; ++nt) {
      int col = nt * 16 + lm;
#pragma unroll
      for (int r = 0; r < 8; ++r)
        G[(mt * 16 + r + 8 * half) * 32 + col] = acc[mt][nt][r];
    }

  float* xr = x + (size_t)b * kXdim;
  for (int i2 = lane; i2 < kD; i2 += 32)
    xr[i2] = dense_out[(size_t)b * kD + i2];
  for (int t = lane; t < 351; t += 32) {
    int r = 0, base = 0;
    while (t >= base + (26 - r)) { base += 26 - r; ++r; }
    int c = r + 1 + (t - base);
    xr[kD + t] = G[r * 32 + c];
  }
  if (lane == 0) xr[kXdim - 1] = 0.f;
}

// ---------------------------------------------------------------------------
// Final layer: out[b] = sigmoid(dot(t2[b], tw3) + tb3). One wave per row.
// ---------------------------------------------------------------------------
__global__ __launch_bounds__(256)
void gemv_sigmoid(const float* __restrict__ t2, const float* __restrict__ w3,
                  const float* __restrict__ b3, float* __restrict__ out) {
  const int lane = threadIdx.x & 31;
  const int b = blockIdx.x * 8 + (threadIdx.x >> 5);
  const float* row = t2 + (size_t)b * 256;
  float s = 0.f;
#pragma unroll
  for (int j = 0; j < 8; ++j)
    s += row[lane + 32 * j] * w3[lane + 32 * j];
  for (int off = 16; off > 0; off >>= 1)
    s += __shfl_down(s, off, 32);
  if (lane == 0)
    out[b] = 1.f / (1.f + __expf(-(s + b3[0])));
}

// ---------------------------------------------------------------------------
extern "C" void kernel_launch(void* const* d_in, const int* in_sizes, int n_in,
                              void* d_out, int out_size, void* d_ws, size_t ws_size,
                              hipStream_t stream) {
  (void)in_sizes; (void)n_in; (void)out_size; (void)ws_size;
  const float* dense_x = (const float*)d_in[0];
  const int*   sparse_x = (const int*)d_in[1];
  // d_in[2] sparse_offsets == arange(B)*P by construction -> bag = l / P
  const float* emb = (const float*)d_in[3];
  const float* bw1 = (const float*)d_in[4];  const float* bb1 = (const float*)d_in[5];
  const float* bw2 = (const float*)d_in[6];  const float* bb2 = (const float*)d_in[7];
  const float* bw3 = (const float*)d_in[8];  const float* bb3 = (const float*)d_in[9];
  const float* tw1 = (const float*)d_in[10]; const float* tb1 = (const float*)d_in[11];
  const float* tw2 = (const float*)d_in[12]; const float* tb2 = (const float*)d_in[13];
  const float* tw3 = (const float*)d_in[14]; const float* tb3 = (const float*)d_in[15];
  float* out = (float*)d_out;

  // ---- workspace carve-up (~105 MB) ----
  char* ws = (char*)d_ws;
  size_t off = 0;
  auto carve = [&](size_t bytes) -> void* {
    void* p = ws + off;
    off = (off + bytes + 255) & ~(size_t)255;
    return p;
  };
  u32* bw1p = (u32*)carve((size_t)32  * 512 * 2);      //  1 K-chunk
  u32* bw2p = (u32*)carve((size_t)512 * 256 * 2);      // 16
  u32* bw3p = (u32*)carve((size_t)256 * 128 * 2);      //  8
  u32* tw1p = (u32*)carve((size_t)480 * 512 * 2);      // 15
  u32* tw2p = (u32*)carve((size_t)512 * 256 * 2);      // 16
  u32* dxp  = (u32*)carve((size_t)kB * 32  * 2);       // packed dense_x (K pad 32)
  u32* h1p  = (u32*)carve((size_t)kB * 512 * 2);       // packed h1 (also t1)
  u32* h2p  = (u32*)carve((size_t)kB * 256 * 2);       // packed h2
  u32* xp   = (u32*)carve((size_t)kB * kXdim * 2);     // packed x
  float* dns = (float*)carve((size_t)kB * kD * 4);     // fp32 dense_out
  float* t2f = (float*)carve((size_t)kB * 256 * 4);    // fp32 t2
  u32*   pld = (u32*)carve((size_t)kB * kNS * 64 * 4); // pooled bf16 pairs
  float* x   = (float*)carve((size_t)kB * kXdim * 4);  // fp32 top-MLP input

  // ---- 1. operand pre-swizzles (one pass each, tiny) ----
  auto packW = [&](const float* W, u32* o, int K, int N, int nKc) {
    int total = nKc * (N >> 4) * 256;
    pack_w<<<(total + 255) / 256, 256, 0, stream>>>(W, o, K, N, nKc);
  };
  packW(bw1, bw1p, kND, 512, 1);
  packW(bw2, bw2p, 512, 256, 16);
  packW(bw3, bw3p, 256, 128, 8);
  packW(tw1, tw1p, 479, 512, 15);
  packW(tw2, tw2p, 512, 256, 16);
  pack_a<<<(kMt * 1 * 256) / 256, 256, 0, stream>>>(dense_x, dxp, kND, kND, 1);

  // ---- 2. bottom MLP (chained layers stay in packed bf16) ----
  gemm_ws<<<dim3(8, 64), 256, 0, stream>>>(dxp, 1,  bw1p, bb1, nullptr, 512, h1p, 16);
  gemm_ws<<<dim3(4, 64), 256, 0, stream>>>(h1p, 16, bw2p, bb2, nullptr, 256, h2p, 8);
  gemm_ws<<<dim3(2, 64), 256, 0, stream>>>(h2p, 8,  bw3p, bb3, dns, kD, nullptr, 0);

  // ---- 3. embedding bag pooling (HBM-bound ~436 MB gather) ----
  emb_pool<<<(kB * kNS) / 8, 256, 0, stream>>>(sparse_x, emb, pld);

  // ---- 4. interactions + top-MLP input assembly, then pack ----
  interact<<<kB / 8, 256, 0, stream>>>(dns, pld, x);
  pack_a<<<(kMt * 15 * 256) / 256, 256, 0, stream>>>(x, xp, kXdim, kXdim, 15);

  // ---- 5. top MLP ----
  gemm_ws<<<dim3(8, 64), 256, 0, stream>>>(xp,  15, tw1p, tb1, nullptr, 512, h1p, 16);
  gemm_ws<<<dim3(4, 64), 256, 0, stream>>>(h1p, 16, tw2p, tb2, t2f, 256, nullptr, 0);
  gemv_sigmoid<<<kB / 8, 256, 0, stream>>>(t2f, tw3, tb3, out);
}
